// Attention_30906584662528
// MI455X (gfx1250) — matrix operand order repaired and checked by reading, not compiled, not run on hardware
//
#include <hip/hip_runtime.h>
#include <hip/hip_bf16.h>
#include <math.h>

// ---------------------------------------------------------------------------
// Problem constants (match the reference)
// ---------------------------------------------------------------------------
#define BB     512
#define TT     128
#define DD     1024
#define HH     1024
#define VV     256
#define STEPS  26
#define H4     4096          // 4*H
#define KC     2048          // D + H (concatenated LSTM input without one-hot)
#define WIHK   1280          // D + V (W_ih K dimension)

typedef __bf16 bf16;
typedef __attribute__((ext_vector_type(16))) __bf16 v16bf;
typedef __attribute__((ext_vector_type(8)))  __bf16 v8bf;
typedef __attribute__((ext_vector_type(4)))  __bf16 v4bf;
typedef __attribute__((ext_vector_type(8)))  float  v8f;
typedef __attribute__((ext_vector_type(4)))  float  v4f;
typedef __attribute__((ext_vector_type(4)))  int    v4i;

#define AS1 __attribute__((address_space(1)))
#define AS3 __attribute__((address_space(3)))

// gfx1250 async global->LDS copy path (guarded: falls back to plain copies)
#if defined(__gfx1250__) && \
    __has_builtin(__builtin_amdgcn_global_load_async_to_lds_b128) && \
    __has_builtin(__builtin_amdgcn_s_wait_asynccnt)
#define USE_ASYNC_LDS 1
#else
#define USE_ASYNC_LDS 0
#endif

__device__ __forceinline__ float fast_tanh(float x) {
#if defined(__gfx1250__) && __has_builtin(__builtin_amdgcn_tanhf)
    return __builtin_amdgcn_tanhf(x);   // v_tanh_f32 (TRANS unit)
#else
    return tanhf(x);
#endif
}

__device__ __forceinline__ float fast_sigmoid(float x) {
    return 1.0f / (1.0f + __expf(-x));
}

// ---------------------------------------------------------------------------
// fp32 -> bf16 cast, 4 elements per thread (all sizes are multiples of 1024)
// ---------------------------------------------------------------------------
__global__ void k_f32_to_bf16_x4(const float* __restrict__ src,
                                 bf16* __restrict__ dst, int n4) {
    int i = blockIdx.x * blockDim.x + threadIdx.x;
    if (i >= n4) return;
    v4f s = *(const v4f*)(src + (size_t)i * 4);
    v4bf d;
#pragma unroll
    for (int k = 0; k < 4; ++k) d[k] = (bf16)s[k];
    *(v4bf*)(dst + (size_t)i * 4) = d;
}

// Build Wcat[4096,2048] = [ W_ih[:, :D] | W_hh ] in bf16, 4 cols per thread
__global__ void k_build_wcat(const float* __restrict__ Wih,
                             const float* __restrict__ Whh,
                             bf16* __restrict__ wcat) {
    int i = blockIdx.x * blockDim.x + threadIdx.x;   // over (H4*KC)/4
    if (i >= (H4 * KC) / 4) return;
    int e = i * 4;
    int n = e >> 11;           // row 0..4095
    int k = e & (KC - 1);      // col, multiple of 4 (never straddles D)
    const float* src = (k < DD) ? (Wih + (size_t)n * WIHK + k)
                                : (Whh + (size_t)n * HH + (k - DD));
    v4f s = *(const v4f*)src;
    v4bf d;
#pragma unroll
    for (int q = 0; q < 4; ++q) d[q] = (bf16)s[q];
    *(v4bf*)(wcat + (size_t)e) = d;
}

__global__ void k_add_bias(const float* __restrict__ a,
                           const float* __restrict__ b,
                           float* __restrict__ o, int n) {
    int i = blockIdx.x * blockDim.x + threadIdx.x;
    if (i < n) o[i] = a[i] + b[i];
}

// ---------------------------------------------------------------------------
// WMMA GEMM:  C[M,N] = A[M,K] * W[N,K]^T (+ bias[N])     (unchanged: codegen
// verified as pipelined global_load_b128 clauses + v_wmma_f32_16x16x32_bf16)
// Wave tile 16x64, block = 8 waves -> 32x256. grid = (N/256, M/32).
// ---------------------------------------------------------------------------
template <typename OutT, bool HAS_BIAS>
__global__ void __launch_bounds__(256)
k_gemm_wmma(const bf16* __restrict__ A, int lda,
            const bf16* __restrict__ W, int ldw,
            OutT* __restrict__ C, int ldc,
            const float* __restrict__ bias,
            int K) {
    const int tid   = threadIdx.x;
    const int wave  = tid >> 5;
    const int lane  = tid & 31;
    const int waveM = wave >> 2;        // 0..1
    const int waveN = wave & 3;         // 0..3
    const int r     = lane & 15;
    const int hs    = lane >> 4;        // K-half select

    const int row0 = blockIdx.y * 32 + waveM * 16;
    const int col0 = blockIdx.x * 256 + waveN * 64;

    v8f zero = {};
    v8f acc[4];
#pragma unroll
    for (int j = 0; j < 4; ++j) acc[j] = zero;

    const bf16* pArow = A + (size_t)(row0 + r) * lda;

    for (int k0 = 0; k0 < K; k0 += 32) {
        const bf16* pA = pArow + k0 + hs * 16;
        __builtin_prefetch((const void*)(pA + 64), 0, 0);   // global_prefetch_b8
        v8bf a0 = *(const v8bf*)(pA);
        v8bf a1 = *(const v8bf*)(pA + 8);
        v16bf af;
#pragma unroll
        for (int i = 0; i < 8; ++i) { af[i] = a0[i]; af[i + 8] = a1[i]; }

#pragma unroll
        for (int j = 0; j < 4; ++j) {
            const bf16* pB = W + (size_t)(col0 + j * 16 + r) * ldw + k0 + hs * 16;
            v16bf bfrag = *(const v16bf*)pB;   // 32 contiguous bytes
            acc[j] = __builtin_amdgcn_wmma_f32_16x16x32_bf16(
                false, af, false, bfrag, (short)0, acc[j], false, false);
        }
    }

#pragma unroll
    for (int j = 0; j < 4; ++j) {
        int n = col0 + j * 16 + r;
        float bv = HAS_BIAS ? bias[n] : 0.0f;
#pragma unroll
        for (int v = 0; v < 8; ++v) {
            int m = row0 + v + hs * 8;
            float val = acc[j][v] + bv;
            C[(size_t)m * ldc + n] = (OutT)val;
        }
    }
}

// ---------------------------------------------------------------------------
// Fused Bahdanau attention, one block per batch row (grid=512, block=256).
// This is the bandwidth-critical kernel (26 x 256KB x 512 streamed), so all
// global traffic moves through 16B-per-lane b128 loads.
//   e[t]    = sum_h tanh(projH[b,t,h] + proj_h[b,h]) * score_w[h]
//   alpha   = softmax_t(e)
//   context = sum_t alpha[t] * batch_H[b,t,:]   -> xcat[b, 0:D] (bf16)
// ---------------------------------------------------------------------------
__global__ void __launch_bounds__(256)
k_attn_fused(const bf16* __restrict__ projH,    // [B,T,H]
             const bf16* __restrict__ projh,    // [B,H]
             const float* __restrict__ score_w, // [H]
             const bf16* __restrict__ bH,       // [B,T,D]
             bf16* __restrict__ xcat) {         // [B,KC], writes [0:D]
    const int b    = blockIdx.x;
    const int tid  = threadIdx.x;
    const int wave = tid >> 5;
    const int lane = tid & 31;

    __shared__ bf16  s_ph[HH];     // proj_h row, bf16 (2KB)
    __shared__ float s_sw[HH];     // score_w; reused as context scratch (4KB)
    __shared__ float s_e[TT];
    __shared__ float s_red[16];

    // ---- stage proj_h row + score_w into LDS ----
#if USE_ASYNC_LDS
    if (tid < 128) {
        bf16* gp = const_cast<bf16*>(projh) + (size_t)b * HH + tid * 8;
        __builtin_amdgcn_global_load_async_to_lds_b128(
            (AS1 v4i*)gp, (AS3 v4i*)(s_ph + tid * 8), 0, 0);
    }
    for (int i = tid; i < HH; i += 256) s_sw[i] = score_w[i];
    if (tid < 128) __builtin_amdgcn_s_wait_asynccnt(0);
    __syncthreads();
#else
    for (int i = tid; i < HH; i += 256) {
        s_ph[i] = projh[(size_t)b * HH + i];
        s_sw[i] = score_w[i];
    }
    __syncthreads();
#endif

    // ---- scores: wave owns t = wave, wave+8, ...; lanes sweep h in v8 ----
    for (int t = wave; t < TT; t += 8) {
        const bf16* row = projH + ((size_t)b * TT + t) * HH;
        float acc = 0.0f;
#pragma unroll
        for (int c = 0; c < 4; ++c) {                 // 4 chunks of 32 lanes*8
            int h0 = c * 256 + lane * 8;
            v8bf pv  = *(const v8bf*)(row + h0);      // global b128
            v8bf phv = *(const v8bf*)(s_ph + h0);     // ds b128
            v4f  swa = *(const v4f*)(s_sw + h0);      // ds b128
            v4f  swb = *(const v4f*)(s_sw + h0 + 4);
#pragma unroll
            for (int i = 0; i < 4; ++i)
                acc += fast_tanh((float)pv[i] + (float)phv[i]) * swa[i];
#pragma unroll
            for (int i = 0; i < 4; ++i)
                acc += fast_tanh((float)pv[4 + i] + (float)phv[4 + i]) * swb[i];
        }
#pragma unroll
        for (int off = 16; off > 0; off >>= 1)
            acc += __shfl_down(acc, off, 32);
        if (lane == 0) s_e[t] = acc;
    }
    __syncthreads();

    // ---- softmax over T = 128 ----
    float ev = (tid < TT) ? s_e[tid] : -3.0e38f;
    float m = ev;
#pragma unroll
    for (int off = 16; off > 0; off >>= 1)
        m = fmaxf(m, __shfl_down(m, off, 32));
    if (lane == 0) s_red[wave] = m;
    __syncthreads();
    if (tid == 0) {
        float mm = s_red[0];
        for (int i = 1; i < 8; ++i) mm = fmaxf(mm, s_red[i]);
        s_red[8] = mm;
    }
    __syncthreads();
    float ex = (tid < TT) ? __expf(ev - s_red[8]) : 0.0f;
    if (tid < TT) s_e[tid] = ex;
    float ssum = ex;
#pragma unroll
    for (int off = 16; off > 0; off >>= 1)
        ssum += __shfl_down(ssum, off, 32);
    if (lane == 0) s_red[wave] = ssum;
    __syncthreads();
    if (tid == 0) {
        float t = 0.0f;
        for (int i = 0; i < 8; ++i) t += s_red[i];
        s_red[9] = 1.0f / t;
    }
    __syncthreads();
    if (tid < TT) s_e[tid] *= s_red[9];
    __syncthreads();

    // ---- context: split-T scheme, 16B-per-lane streaming of batch_H[b] ----
    // tid[7]   selects t-half (0..63 / 64..127)
    // tid[6:0] selects an 8-wide contiguous d slice (128*8 = 1024)
    const int g  = tid >> 7;
    const int dt = (tid & 127) * 8;
    float acc[8];
#pragma unroll
    for (int i = 0; i < 8; ++i) acc[i] = 0.0f;

    const bf16* base = bH + ((size_t)b * TT + g * 64) * DD + dt;
    for (int t = 0; t < 64; ++t) {
        float a = s_e[g * 64 + t];
        v8bf hv = *(const v8bf*)(base + (size_t)t * DD);  // global b128
#pragma unroll
        for (int i = 0; i < 8; ++i) acc[i] += a * (float)hv[i];
    }

    // combine the two t-halves through LDS (reuse s_sw as scratch)
    if (g == 0) {
#pragma unroll
        for (int i = 0; i < 8; ++i) s_sw[dt + i] = acc[i];
    }
    __syncthreads();
    if (g == 1) {
        v8bf outv;
#pragma unroll
        for (int i = 0; i < 8; ++i) outv[i] = (bf16)(acc[i] + s_sw[dt + i]);
        *(v8bf*)(xcat + (size_t)b * KC + dt) = outv;      // 16B store
    }
}

// ---------------------------------------------------------------------------
// LSTM pointwise: add one-hot column of W_ih, apply gates, update c,
// write h (bf16) into xcat[:, D:] and out_hid[:, s, :].
// Vectorized: each thread owns 4 consecutive h (b128 gate loads).
// ---------------------------------------------------------------------------
__global__ void __launch_bounds__(256)
k_lstm_pointwise(const float* __restrict__ gates,    // [B,4H]
                 const float* __restrict__ Wih,      // [4H, D+V] fp32
                 const long long* __restrict__ text, // [B,STEPS]
                 int s,
                 float* __restrict__ c,              // [B,H]
                 bf16* __restrict__ xcat,            // [B,KC], h at +D
                 bf16* __restrict__ outhid) {        // [(b*STEPS+s)*H]
    const int b   = blockIdx.x;
    const int tid = threadIdx.x;
    const int tok = (int)text[(size_t)b * STEPS + s];
    const size_t g0 = (size_t)b * H4;
    const int j = tid * 4;                            // 256*4 = 1024 = H

    v4f ig = *(const v4f*)(gates + g0 + j);
    v4f fg = *(const v4f*)(gates + g0 + HH + j);
    v4f gg = *(const v4f*)(gates + g0 + 2 * HH + j);
    v4f og = *(const v4f*)(gates + g0 + 3 * HH + j);
    v4f cv = *(const v4f*)(c + (size_t)b * HH + j);

    v4f cn;
    v4bf hn;
#pragma unroll
    for (int q = 0; q < 4; ++q) {
        // one-hot contribution = gathered column (D + tok) of W_ih
        float wi = Wih[(size_t)(j + q)            * WIHK + DD + tok];
        float wf = Wih[(size_t)(HH + j + q)       * WIHK + DD + tok];
        float wg = Wih[(size_t)(2 * HH + j + q)   * WIHK + DD + tok];
        float wo = Wih[(size_t)(3 * HH + j + q)   * WIHK + DD + tok];
        float si = fast_sigmoid(ig[q] + wi);
        float sf = fast_sigmoid(fg[q] + wf);
        float so = fast_sigmoid(og[q] + wo);
        float cq = sf * cv[q] + si * fast_tanh(gg[q] + wg);
        cn[q] = cq;
        hn[q] = (bf16)(so * fast_tanh(cq));
    }

    *(v4f*)(c + (size_t)b * HH + j) = cn;
    *(v4bf*)(xcat + (size_t)b * KC + DD + j) = hn;
    *(v4bf*)(outhid + ((size_t)b * STEPS + s) * HH + j) = hn;
}

// ---------------------------------------------------------------------------
// Launch
// ---------------------------------------------------------------------------
extern "C" void kernel_launch(void* const* d_in, const int* in_sizes, int n_in,
                              void* d_out, int out_size, void* d_ws, size_t ws_size,
                              hipStream_t stream) {
    (void)in_sizes; (void)n_in; (void)out_size; (void)ws_size;

    const float*      batch_H = (const float*)d_in[0];
    const long long*  text    = (const long long*)d_in[1];
    const float*      i2h_w   = (const float*)d_in[2];
    const float*      h2h_w   = (const float*)d_in[3];
    const float*      h2h_b   = (const float*)d_in[4];
    const float*      score_w = (const float*)d_in[5];
    const float*      W_ih    = (const float*)d_in[6];
    const float*      W_hh    = (const float*)d_in[7];
    const float*      b_ih    = (const float*)d_in[8];
    const float*      b_hh    = (const float*)d_in[9];
    const float*      gen_w   = (const float*)d_in[10];
    const float*      gen_b   = (const float*)d_in[11];
    float*            out     = (float*)d_out;

    // ---- workspace carve-up (~316 MB total) ----
    char* w = (char*)d_ws;
    auto alloc = [&](size_t bytes) {
        char* p = w;
        w += (bytes + 255) & ~(size_t)255;
        return p;
    };
    bf16*  bHbf    = (bf16*)alloc((size_t)BB * TT * DD * 2);   // 128 MB
    bf16*  projH   = (bf16*)alloc((size_t)BB * TT * HH * 2);   // 128 MB
    bf16*  w_i2h   = (bf16*)alloc((size_t)HH * DD * 2);
    bf16*  w_h2h   = (bf16*)alloc((size_t)HH * HH * 2);
    bf16*  w_cat   = (bf16*)alloc((size_t)H4 * KC * 2);        // 16 MB
    bf16*  w_gen   = (bf16*)alloc((size_t)VV * HH * 2);
    float* bias4h  = (float*)alloc((size_t)H4 * 4);
    bf16*  xcat    = (bf16*)alloc((size_t)BB * KC * 2);        // [context|h]
    bf16*  projh_b = (bf16*)alloc((size_t)BB * HH * 2);
    float* c_state = (float*)alloc((size_t)BB * HH * 4);
    float* gates   = (float*)alloc((size_t)BB * H4 * 4);
    bf16*  outhid  = (bf16*)alloc((size_t)BB * STEPS * HH * 2);

    // ---- one-time precompute ----
    {
        int n4 = (BB * TT * DD) / 4;
        k_f32_to_bf16_x4<<<(n4 + 255) / 256, 256, 0, stream>>>(batch_H, bHbf, n4);
        n4 = (HH * DD) / 4;
        k_f32_to_bf16_x4<<<(n4 + 255) / 256, 256, 0, stream>>>(i2h_w, w_i2h, n4);
        n4 = (HH * HH) / 4;
        k_f32_to_bf16_x4<<<(n4 + 255) / 256, 256, 0, stream>>>(h2h_w, w_h2h, n4);
        n4 = (VV * HH) / 4;
        k_f32_to_bf16_x4<<<(n4 + 255) / 256, 256, 0, stream>>>(gen_w, w_gen, n4);
        n4 = (H4 * KC) / 4;
        k_build_wcat<<<(n4 + 255) / 256, 256, 0, stream>>>(W_ih, W_hh, w_cat);
        k_add_bias<<<(H4 + 255) / 256, 256, 0, stream>>>(b_ih, b_hh, bias4h, H4);
    }
    (void)hipMemsetAsync(xcat,    0, (size_t)BB * KC * 2, stream);   // h0 = 0
    (void)hipMemsetAsync(c_state, 0, (size_t)BB * HH * 4, stream);   // c0 = 0

    // ---- proj_H = batch_H @ i2h_w^T   (65536 x 1024 x 1024) ----
    k_gemm_wmma<bf16, false><<<dim3(HH / 256, (BB * TT) / 32), 256, 0, stream>>>(
        bHbf, DD, w_i2h, DD, projH, HH, nullptr, DD);

    // ---- 26-step recurrence ----
    for (int s = 0; s < STEPS; ++s) {
        // proj_h = h @ h2h_w^T + h2h_b   (A = h part of xcat, lda = KC)
        k_gemm_wmma<bf16, true><<<dim3(HH / 256, BB / 32), 256, 0, stream>>>(
            xcat + DD, KC, w_h2h, HH, projh_b, HH, h2h_b, HH);

        // attention: scores + softmax + context -> xcat[:, 0:D]
        k_attn_fused<<<BB, 256, 0, stream>>>(projH, projh_b, score_w, bHbf, xcat);

        // gates = [context|h] @ [W_ih[:, :D]|W_hh]^T + (b_ih + b_hh)
        k_gemm_wmma<float, true><<<dim3(H4 / 256, BB / 32), 256, 0, stream>>>(
            xcat, KC, w_cat, KC, gates, H4, bias4h, KC);

        // pointwise LSTM (adds the one-hot column of W_ih)
        k_lstm_pointwise<<<BB, 256, 0, stream>>>(gates, W_ih, text, s,
                                                 c_state, xcat, outhid);
    }

    // ---- logits = out_hid @ gen_w^T + gen_b   (13312 x 256 x 1024) ----
    k_gemm_wmma<float, true><<<dim3(VV / 256, (BB * STEPS) / 32), 256, 0, stream>>>(
        outhid, HH, w_gen, HH, out, VV, gen_b, HH);
}
